// SAGTGraphConstructor_47614007443565
// MI455X (gfx1250) — compile-verified, matrix operand order checked by reading, not run on hardware
//
#include <hip/hip_runtime.h>
#include <hip/hip_fp16.h>
#include <hip/hip_bf16.h>

// ---------------- constants from the reference ----------------
#define Nn    1024
#define Tt    96
#define Bb    16
#define Cc    2
#define KCAND 20
#define Ee    (Nn * KCAND)   // 20480 candidate edges
#define NEV   16             // subspace size (= SPEC_RANK)
#define NMFR  8              // ROLE_DIM / RANK_LR
#define EPSf  1e-8f

typedef __attribute__((ext_vector_type(16))) _Float16 v16h;
typedef __attribute__((ext_vector_type(8)))  float    v8f;

// CDNA5 async global->LDS path, guarded so either toolchain compiles.
#if defined(__has_builtin)
#  if __has_builtin(__builtin_amdgcn_global_load_async_to_lds_b128) && \
      __has_builtin(__builtin_amdgcn_s_wait_asynccnt)
#    define USE_ASYNC_LDS 1
#  endif
#endif
#ifndef USE_ASYNC_LDS
#  define USE_ASYNC_LDS 0
#endif
#if USE_ASYNC_LDS
typedef int v4i_t __attribute__((ext_vector_type(4)));
typedef __attribute__((address_space(1))) v4i_t gv4i;   // global int4
typedef __attribute__((address_space(3))) v4i_t lv4i;   // LDS int4
#endif

// ---------------- helpers ----------------
__device__ __forceinline__ float breduce_sum(float v, float* red) {
  int t = threadIdx.x;
  red[t] = v; __syncthreads();
  for (int s = blockDim.x >> 1; s > 0; s >>= 1) {
    if (t < s) red[t] += red[t + s];
    __syncthreads();
  }
  float r = red[0]; __syncthreads();
  return r;
}

__device__ __forceinline__ v8f wmma_f16(v16h a, v16h b, v8f c) {
  return __builtin_amdgcn_wmma_f32_16x16x32_f16(
      false, a, false, b, (short)0, c, false, false);
}

// A-fragment gather: rows row0..row0+15, K = k0..k0+31, row-major with given
// stride. CDNA5 16-bit A layout: lane L holds row M=L&15; lanes 0-15 carry
// K in {0..7,16..23}, lanes 16-31 carry K in {8..15,24..31}.
// For B = X^T (as in C = X X^T) the same gather (row = output column) is the
// matching B-fragment.
__device__ __forceinline__ v16h load_fragA(const _Float16* X, int row0, int k0, int stride) {
  int lane = threadIdx.x & 31;
  int m = lane & 15, hi = lane >> 4;
  const _Float16* p = X + (size_t)(row0 + m) * stride + k0 + hi * 8;
  v16h f;
#pragma unroll
  for (int h = 0; h < 8; ++h) { f[h] = p[h]; f[h + 8] = p[16 + h]; }
  return f;
}

// B-fragment gather from a row-major [K x 16] f16 panel: lane holds column
// N=L&15; halves iterate K with the same hi-split as the A layout.
__device__ __forceinline__ v16h load_fragN(const _Float16* X, int k0) {
  int lane = threadIdx.x & 31;
  int n = lane & 15, hi = lane >> 4;
  const _Float16* p = X + (size_t)(k0 + hi * 8) * 16 + n;
  v16h f;
#pragma unroll
  for (int h = 0; h < 8; ++h) { f[h] = p[(size_t)h * 16]; f[h + 8] = p[(size_t)(16 + h) * 16]; }
  return f;
}

// ---------------- K0: init atomics ----------------
__global__ void k_init(int* gmax) {
  if (threadIdx.x < 2) gmax[threadIdx.x] = 0;
}

// ---------------- K1: series prep ----------------
__global__ __launch_bounds__(128) void k_prep(const float* __restrict__ x,
                                              float* s, float* sn,
                                              float* mu, float* sd, float* nrm,
                                              _Float16* snh, _Float16* ph, _Float16* qh) {
  __shared__ float red[128];
  int n = blockIdx.x, t = threadIdx.x;
  float sv = 0.f;
  if (t < Tt) {
#pragma unroll
    for (int b = 0; b < Bb; ++b)
#pragma unroll
      for (int c = 0; c < Cc; ++c)
        sv += x[(((size_t)b * Nn + n) * Tt + t) * Cc + c];
    sv *= (1.0f / (Bb * Cc));
    s[(size_t)n * Tt + t] = sv;
  }
  float sum1 = breduce_sum((t < Tt) ? sv : 0.f, red);
  float sum2 = breduce_sum((t < Tt) ? sv * sv : 0.f, red);
  float m    = sum1 / Tt;
  float var  = sum2 / Tt - m * m;
  float sdev = sqrtf(fmaxf(var, 0.f));
  if (t == 0) { mu[n] = m; sd[n] = sdev; nrm[n] = sqrtf(fmaxf(sum2, 0.f)); }
  if (t < Tt) {
    float z = (sv - m) / (sdev + EPSf);
    sn[(size_t)n * Tt + t]  = z;
    snh[(size_t)n * Tt + t] = (_Float16)z;
  }
  __syncthreads();
  if (t < Tt) {
    float z = sn[(size_t)n * Tt + t];
    ph[(size_t)n * Tt + t] = (t < Tt - 1) ? (_Float16)z : (_Float16)0.f;
    qh[(size_t)n * Tt + t] = (t < Tt - 1) ? (_Float16)sn[(size_t)n * Tt + t + 1]
                                          : (_Float16)0.f;
  }
}

// ---------------- K2: WMMA relation matrix (64x32 super-tile, LDS staged) ----
// Per WG: stage i-panels (64x96) and j-panels (32x96) of snh/ph/qh into LDS
// (async-to-LDS when available), then 8 waves each compute one 16x16 tile:
//   c0 = sn.sn^T, X = P.Q^T, Y = Q.P^T; R = 0.5|c0/T| + 0.25|X+Y|/(T-1).
__global__ __launch_bounds__(256) void k_relation(const _Float16* __restrict__ snh,
                                                  const _Float16* __restrict__ ph,
                                                  const _Float16* __restrict__ qh,
                                                  float* __restrict__ R,
                                                  _Float16* __restrict__ Rh) {
  __shared__ __align__(16) _Float16 Pi[3][64 * Tt];  // 36 KB
  __shared__ __align__(16) _Float16 Pj[3][32 * Tt];  // 18 KB
  int i0 = blockIdx.y * 64;
  int j0 = blockIdx.x * 32;
  int t = threadIdx.x;
  const _Float16* gsrc[3] = { snh, ph, qh };
#pragma unroll
  for (int pnl = 0; pnl < 3; ++pnl) {
    const char* gi = (const char*)(gsrc[pnl] + (size_t)i0 * Tt);
    const char* gj = (const char*)(gsrc[pnl] + (size_t)j0 * Tt);
    char* li = (char*)&Pi[pnl][0];
    char* lj = (char*)&Pj[pnl][0];
#if USE_ASYNC_LDS
    for (int off = t * 16; off < 64 * Tt * 2; off += 256 * 16)
      __builtin_amdgcn_global_load_async_to_lds_b128(
          (gv4i*)(gi + off), (lv4i*)(li + off), 0, 0);
    for (int off = t * 16; off < 32 * Tt * 2; off += 256 * 16)
      __builtin_amdgcn_global_load_async_to_lds_b128(
          (gv4i*)(gj + off), (lv4i*)(lj + off), 0, 0);
#else
    for (int off = t * 16; off < 64 * Tt * 2; off += 256 * 16)
      *(float4*)(li + off) = *(const float4*)(gi + off);
    for (int off = t * 16; off < 32 * Tt * 2; off += 256 * 16)
      *(float4*)(lj + off) = *(const float4*)(gj + off);
#endif
  }
#if USE_ASYNC_LDS
  __builtin_amdgcn_s_wait_asynccnt(0);
#endif
  __syncthreads();

  int w  = t >> 5;
  int il = (w & 3) * 16;   // local tile row within 64
  int jl = (w >> 2) * 16;  // local tile col within 32
  v8f c0 = {}, xx = {}, yy = {};
#pragma unroll
  for (int k0 = 0; k0 < Tt; k0 += 32) {
    c0 = wmma_f16(load_fragA(&Pi[0][0], il, k0, Tt), load_fragA(&Pj[0][0], jl, k0, Tt), c0);
    xx = wmma_f16(load_fragA(&Pi[1][0], il, k0, Tt), load_fragA(&Pj[2][0], jl, k0, Tt), xx);
    yy = wmma_f16(load_fragA(&Pi[2][0], il, k0, Tt), load_fragA(&Pj[1][0], jl, k0, Tt), yy);
  }
  int lane = t & 31, nc = lane & 15, hi = lane >> 4;
#pragma unroll
  for (int v = 0; v < 8; ++v) {
    int gr = i0 + il + v + 8 * hi;
    int gc = j0 + jl + nc;
    float r = 0.5f * fabsf(c0[v] * (1.0f / Tt))
            + 0.25f * fabsf((xx[v] + yy[v]) * (1.0f / (Tt - 1)));
    R[(size_t)gr * Nn + gc]  = r;
    Rh[(size_t)gr * Nn + gc] = (_Float16)r;
  }
}

// ---------------- K3: per-row top-20 candidates ----------------
__global__ __launch_bounds__(32) void k_topk(const float* __restrict__ R, int* cols) {
  int row = blockIdx.x, lane = threadIdx.x;
  const float* r = R + (size_t)row * Nn;
  float va[32];
#pragma unroll
  for (int g = 0; g < 32; ++g) {
    int c = g * 32 + lane;
    va[g] = (c == row) ? -1e30f : r[c];
  }
  for (int k = 0; k < KCAND; ++k) {
    float bv = -1e30f; int bc = 0;
#pragma unroll
    for (int g = 0; g < 32; ++g) {
      int c = g * 32 + lane;
      if (va[g] > bv) { bv = va[g]; bc = c; }
    }
#pragma unroll
    for (int off = 16; off; off >>= 1) {
      float ov = __shfl_xor(bv, off, 32);
      int   oc = __shfl_xor(bc, off, 32);
      if (ov > bv || (ov == bv && oc < bc)) { bv = ov; bc = oc; }
    }
    if ((bc & 31) == lane) va[bc >> 5] = -1e30f;
    if (lane == 0) cols[row * KCAND + k] = bc;
  }
}

// ---------------- eigen phase: multi-WG WMMA subspace iteration ----------------
// K4a: deterministic pseudo-random init of V (f32 + f16).
__global__ void k_einit(float* Va, _Float16* Vah) {
  int i = blockIdx.x * blockDim.x + threadIdx.x;
  if (i >= Nn) return;
#pragma unroll
  for (int k = 0; k < NEV; ++k) {
    unsigned u = (unsigned)(i * NEV + k + 1) * 2654435761u;
    float val = ((u >> 9) & 0xFFFFFF) * (1.0f / 16777216.0f) + 1e-3f;
    Va[(size_t)i * NEV + k]  = val;
    Vah[(size_t)i * NEV + k] = (_Float16)val;
  }
}

// K4b: Vout[1024x16] = M(f16) * V(f16) via WMMA; one wave per 16-row block.
__global__ __launch_bounds__(32) void k_mv(const _Float16* __restrict__ Rh,
                                           const _Float16* __restrict__ Vh,
                                           float* __restrict__ Vout) {
  int i0 = blockIdx.x * 16;
  v8f acc = {};
  for (int k0 = 0; k0 < Nn; k0 += 32) {
    __builtin_prefetch(Rh + (size_t)i0 * Nn + k0 + 64, 0, 0);
    acc = wmma_f16(load_fragA(Rh, i0, k0, Nn), load_fragN(Vh, k0), acc);
  }
  int lane = threadIdx.x & 31, n = lane & 15, hi = lane >> 4;
#pragma unroll
  for (int v = 0; v < 8; ++v)
    Vout[(size_t)(i0 + v + 8 * hi) * NEV + n] = acc[v];
}

// K4c: G = V^T V (16x16) then Cholesky G = L L^T (thread 0) -> Lc.
__global__ __launch_bounds__(256) void k_gram(const float* __restrict__ V, float* Lc) {
  __shared__ float Gs[NEV * NEV];
  int t = threadIdx.x;
  if (t < NEV * NEV) {
    int a = t >> 4, b = t & 15;
    float s = 0.f;
    for (int j = 0; j < Nn; ++j)
      s += V[(size_t)j * NEV + a] * V[(size_t)j * NEV + b];
    Gs[t] = s;
  }
  __syncthreads();
  if (t == 0) {
    float L[NEV][NEV];
    for (int a = 0; a < NEV; ++a)
      for (int b = 0; b < NEV; ++b) L[a][b] = 0.f;
    for (int a = 0; a < NEV; ++a) {
      float d = Gs[a * NEV + a];
      for (int j = 0; j < a; ++j) d -= L[a][j] * L[a][j];
      d = sqrtf(fmaxf(d, 1e-20f));
      L[a][a] = d;
      for (int b = a + 1; b < NEV; ++b) {
        float s2 = Gs[b * NEV + a];
        for (int j = 0; j < a; ++j) s2 -= L[b][j] * L[a][j];
        L[b][a] = s2 / d;
      }
    }
    for (int a = 0; a < NEV; ++a)
      for (int b = 0; b < NEV; ++b) Lc[a * NEV + b] = L[a][b];
  }
}

// K4d: Q = V * L^-T (Cholesky QR), write f32 + f16.
__global__ void k_ortho(const float* __restrict__ Vin, const float* __restrict__ Lc,
                        float* Vout, _Float16* Vhout) {
  int i = blockIdx.x * blockDim.x + threadIdx.x;
  if (i >= Nn) return;
  float v[NEV], q[NEV];
#pragma unroll
  for (int k = 0; k < NEV; ++k) v[k] = Vin[(size_t)i * NEV + k];
#pragma unroll
  for (int k = 0; k < NEV; ++k) {
    float s = v[k];
    for (int j = 0; j < k; ++j) s -= q[j] * Lc[k * NEV + j];
    q[k] = s / Lc[k * NEV + k];
  }
#pragma unroll
  for (int k = 0; k < NEV; ++k) {
    Vout[(size_t)i * NEV + k]  = q[k];
    Vhout[(size_t)i * NEV + k] = (_Float16)q[k];
  }
}

// K4e: Ritz values + spectral signature tail.
__global__ __launch_bounds__(1024) void k_lam(const float* __restrict__ V,
                                              const float* __restrict__ AV,
                                              const float* __restrict__ srcsig,
                                              float* rho, float* lam8,
                                              int* idx8, int* idxw, float* wscale) {
  __shared__ float red[1024];
  __shared__ float lam[NEV];
  int i = threadIdx.x;
  for (int k = 0; k < NEV; ++k) {
    float lk = breduce_sum(V[(size_t)i * NEV + k] * AV[(size_t)i * NEV + k], red);
    if (i == 0) lam[k] = lk;
  }
  __syncthreads();
  if (i == 0) {
    int ord[NEV], orda[NEV];
    for (int k = 0; k < NEV; ++k) { ord[k] = k; orda[k] = k; }
    for (int a = 0; a < NEV; ++a)
      for (int b = a + 1; b < NEV; ++b)
        if (fabsf(lam[ord[b]]) > fabsf(lam[ord[a]])) { int tt = ord[a]; ord[a] = ord[b]; ord[b] = tt; }
    for (int a = 0; a < NEV; ++a)
      for (int b = a + 1; b < NEV; ++b)
        if (lam[orda[b]] > lam[orda[a]]) { int tt = orda[a]; orda[a] = orda[b]; orda[b] = tt; }
    float top[NEV];
    for (int r = 0; r < NEV; ++r) top[r] = fabsf(lam[ord[r]]);
    float t0 = top[0] + EPSf;
    for (int r = 0; r < NEV; ++r) top[r] /= t0;
    float sig[4];
    for (int m = 0; m < 4; ++m) {
      float acc = 0.f;
      for (int r = 0; r < NEV; ++r) {
        float pw = top[r];
        for (int q = 0; q < m; ++q) pw *= top[r];
        acc += pw;
      }
      sig[m] = acc / NEV;
    }
    float dist[3], dmin = 1e30f;
    for (int m = 0; m < 3; ++m) {
      float d = 0.f;
      for (int q = 0; q < 4; ++q) {
        float df = srcsig[m * 4 + q] - sig[q];
        d += df * df;
      }
      dist[m] = d; dmin = fminf(dmin, d);
    }
    float es = 0.f, ev[3];
    for (int m = 0; m < 3; ++m) { ev[m] = expf(-(dist[m] - dmin)); es += ev[m]; }
    for (int m = 0; m < 3; ++m) rho[m] = ev[m] / es;
    for (int r = 0; r < NMFR; ++r) {
      idx8[r]   = ord[r];
      lam8[r]   = lam[ord[r]];
      idxw[r]   = orda[r];
      wscale[r] = sqrtf(fabsf(lam[orda[r]]));
    }
  }
}

// K4f: E8 (rank-8 eigvecs) + NMF init W0 = |v|*sqrt(|lam|) (f32 + f16 padded).
__global__ void k_post(const float* __restrict__ Va, const int* __restrict__ idx8,
                       const int* __restrict__ idxw, const float* __restrict__ wscale,
                       float* E8, float* Wm, _Float16* Wh) {
  int i = blockIdx.x * blockDim.x + threadIdx.x;
  if (i >= Nn) return;
#pragma unroll
  for (int r = 0; r < NMFR; ++r) {
    E8[(size_t)i * NMFR + r] = Va[(size_t)i * NEV + idx8[r]];
    float w0 = fabsf(Va[(size_t)i * NEV + idxw[r]]) * wscale[r];
    Wm[(size_t)i * NMFR + r] = w0;
    Wh[(size_t)i * NEV + r]  = (_Float16)w0;
    Wh[(size_t)i * NEV + r + NMFR] = (_Float16)0.f;   // zero pad cols 8..15
  }
}

// ---------------- NMF phase: per-iteration multi-WG WMMA ----------------
// K5a: C = W^T W (8x8), 64 serial entries: deterministic.
__global__ __launch_bounds__(64) void k_nmf_gram(const float* __restrict__ Wm, float* C) {
  int t = threadIdx.x;
  int a = t >> 3, b = t & 7;
  float s = 0.f;
  for (int j = 0; j < Nn; ++j)
    s += Wm[(size_t)j * NMFR + a] * Wm[(size_t)j * NMFR + b];
  C[t] = s;
}

// K5b: one GD step. mw = M*W via WMMA; W' = max(W - 0.2*(W C - mw), 0).
__global__ __launch_bounds__(32) void k_nmf_step(const _Float16* __restrict__ Rh,
                                                 const _Float16* __restrict__ Wh,
                                                 const float* __restrict__ Wm,
                                                 const float* __restrict__ C,
                                                 float* Wm2, _Float16* Wh2) {
  int i0 = blockIdx.x * 16;
  v8f acc = {};
  for (int k0 = 0; k0 < Nn; k0 += 32)
    acc = wmma_f16(load_fragA(Rh, i0, k0, Nn), load_fragN(Wh, k0), acc);
  int lane = threadIdx.x & 31, n = lane & 15, hi = lane >> 4;
#pragma unroll
  for (int v = 0; v < 8; ++v) {
    int row = i0 + v + 8 * hi;
    if (n < NMFR) {
      float wrow[NMFR];
#pragma unroll
      for (int k = 0; k < NMFR; ++k) wrow[k] = Wm[(size_t)row * NMFR + k];
      float wc = 0.f;
#pragma unroll
      for (int k = 0; k < NMFR; ++k) wc += wrow[k] * C[k * NMFR + n];
      float wn = fmaxf(wrow[n] - 0.2f * (wc - acc[v]), 0.f);   // lr*4 = 0.2
      Wm2[(size_t)row * NMFR + n] = wn;
      Wh2[(size_t)row * NEV + n]  = (_Float16)wn;
    } else {
      Wh2[(size_t)row * NEV + n] = (_Float16)0.f;
    }
  }
}

// ---------------- K6: U = W * Beff (Beff = sum_m rho_m B_m) ----------------
__global__ void k_roleU(const float* __restrict__ Wm, const float* __restrict__ rho,
                        const float* __restrict__ srcB, float* U) {
  int idx = blockIdx.x * blockDim.x + threadIdx.x;
  if (idx >= Nn * NMFR) return;
  int i = idx >> 3, r = idx & 7;
  float u = 0.f;
#pragma unroll
  for (int k = 0; k < NMFR; ++k) {
    float be = 0.f;
#pragma unroll
    for (int m = 0; m < 3; ++m) be += rho[m] * srcB[(m * NMFR + k) * NMFR + r];
    u += Wm[(size_t)i * NMFR + k] * be;
  }
  U[idx] = u;
}

// ---------------- K7: global max |X_i . Y_j| over i!=j (order-independent) ----
__global__ __launch_bounds__(256) void k_pairmax(const float* __restrict__ X,
                                                 const float* __restrict__ Y,
                                                 const float* __restrict__ scale,
                                                 int* gmax, int slot) {
  __shared__ float red[256];
  int p = blockIdx.x * 256 + threadIdx.x;
  int i = p >> 10, j = p & (Nn - 1);
  float v = 0.f;
  if (i != j) {
    float d = 0.f;
#pragma unroll
    for (int r = 0; r < NMFR; ++r) {
      float sc = scale ? scale[r] : 1.0f;
      d += sc * X[(size_t)i * NMFR + r] * Y[(size_t)j * NMFR + r];
    }
    v = fabsf(d);
  }
  int t = threadIdx.x;
  red[t] = v; __syncthreads();
  for (int s = 128; s > 0; s >>= 1) {
    if (t < s) red[t] = fmaxf(red[t], red[t + s]);
    __syncthreads();
  }
  if (t == 0) atomicMax(gmax + slot, __float_as_int(red[0]));  // non-negative floats
}

// ---------------- K8: low-rank branch score (masked row softmax) -------------
__global__ __launch_bounds__(32) void k_lrscore(const float* __restrict__ E8,
                                                const float* __restrict__ lam8,
                                                const int* __restrict__ cols,
                                                const int* __restrict__ gmax,
                                                float* sc_lr) {
  int row = blockIdx.x, lane = threadIdx.x;
  float inv = 1.0f / (__int_as_float(gmax[0]) + EPSf);
  float li[NMFR];
#pragma unroll
  for (int r = 0; r < NMFR; ++r) li[r] = lam8[r] * E8[(size_t)row * NMFR + r];
  float dsum = 0.f;
  for (int j = lane; j < Nn; j += 32) {
    if (j == row) continue;
    float d = 0.f;
#pragma unroll
    for (int r = 0; r < NMFR; ++r) d += li[r] * E8[(size_t)j * NMFR + r];
    float a = fabsf(d);
    if (a > 0.f) dsum += expf(a * inv);
  }
#pragma unroll
  for (int off = 16; off; off >>= 1) dsum += __shfl_xor(dsum, off, 32);
  if (lane < KCAND) {
    int c = cols[row * KCAND + lane];
    float d = 0.f;
#pragma unroll
    for (int r = 0; r < NMFR; ++r) d += li[r] * E8[(size_t)c * NMFR + r];
    float a = fabsf(d);
    sc_lr[row * KCAND + lane] = (a > 0.f) ? expf(a * inv) / dsum : 0.f;
  }
}

// ---------------- K9: role branch score ----------------
__global__ void k_rolescore(const float* __restrict__ U, const float* __restrict__ Wm,
                            const int* __restrict__ cols, const int* __restrict__ gmax,
                            float* sc) {
  int e = blockIdx.x * blockDim.x + threadIdx.x;
  if (e >= Ee) return;
  int row = e / KCAND, c = cols[e];
  float d = 0.f;
#pragma unroll
  for (int r = 0; r < NMFR; ++r) d += U[(size_t)row * NMFR + r] * Wm[(size_t)c * NMFR + r];
  sc[e] = fabsf(d) / (__int_as_float(gmax[1]) + EPSf);
}

// ---------------- K10: residual MLP branch ----------------
__global__ void k_resscore(const float* __restrict__ s, const float* __restrict__ sn,
                           const float* __restrict__ mu, const float* __restrict__ sd,
                           const float* __restrict__ nrm, const int* __restrict__ cols,
                           const float* __restrict__ w1, const float* __restrict__ b1,
                           const float* __restrict__ w2, const float* __restrict__ b2,
                           float* sc) {
  int e = blockIdx.x * blockDim.x + threadIdx.x;
  if (e >= Ee) return;
  int r = e / KCAND, c = cols[e];
  const float* ar = sn + (size_t)r * Tt;
  const float* ac = sn + (size_t)c * Tt;
  const float* xr = s  + (size_t)r * Tt;
  const float* xc = s  + (size_t)c * Tt;
  float dot0 = 0.f, dl = 0.f, dlT = 0.f, dr = 0.f;
  for (int t = 0; t < Tt; ++t) { dot0 += ar[t] * ac[t]; dr += xr[t] * xc[t]; }
  for (int t = 0; t < Tt - 1; ++t) { dl += ar[t] * ac[t + 1]; dlT += ac[t] * ar[t + 1]; }
  float f[8] = { dot0 / Tt, dl / (Tt - 1), dlT / (Tt - 1),
                 dr / (nrm[r] * nrm[c] + EPSf),
                 mu[r], mu[c], sd[r], sd[c] };
  float out = b2[0];
#pragma unroll
  for (int h = 0; h < 32; ++h) {
    float a = b1[h];
#pragma unroll
    for (int k = 0; k < 8; ++k) a += f[k] * w1[k * 32 + h];
    out += fmaxf(a, 0.f) * w2[h];
  }
  sc[e] = out;
}

// ---------------- K11: mean/std of the three score vectors -------------------
__global__ __launch_bounds__(1024) void k_stats(const float* a, const float* b,
                                                const float* c, float* stats) {
  __shared__ float red[1024];
  __shared__ float tots[6];
  int t = threadIdx.x;
  float s[6] = {0, 0, 0, 0, 0, 0};
  for (int e = t; e < Ee; e += 1024) {
    float x = a[e]; s[0] += x; s[1] += x * x;
    x = b[e];       s[2] += x; s[3] += x * x;
    x = c[e];       s[4] += x; s[5] += x * x;
  }
  for (int q = 0; q < 6; ++q) {
    float tv = breduce_sum(s[q], red);
    if (t == 0) tots[q] = tv;
  }
  __syncthreads();
  if (t == 0) {
    for (int p = 0; p < 3; ++p) {
      float m = tots[2 * p] / Ee;
      float var = tots[2 * p + 1] / Ee - m * m;
      stats[2 * p] = m;
      stats[2 * p + 1] = sqrtf(fmaxf(var, 0.f));
    }
  }
}

// ---------------- K12: combine + sigmoid ----------------
__global__ void k_final(const float* lr, const float* ro, const float* re,
                        const float* st, float* sig) {
  int e = blockIdx.x * blockDim.x + threadIdx.x;
  if (e >= Ee) return;
  float z = 0.3f * (lr[e] - st[0]) / (st[1] + EPSf)
          + 0.4f * (ro[e] - st[2]) / (st[3] + EPSf)
          + 0.1f * (re[e] - st[4]) / (st[5] + EPSf);
  sig[e] = 1.0f / (1.0f + expf(-z));
}

// ---------------- K13: row softmax + broadcast to [B,N,N] --------------------
__global__ __launch_bounds__(256) void k_emit(const float* __restrict__ sig,
                                              const int* __restrict__ cols,
                                              float* __restrict__ out) {
  __shared__ __align__(16) float rowbuf[Nn];
  __shared__ float evv[KCAND];
  __shared__ float sinv;
  int row = blockIdx.x, t = threadIdx.x;
  for (int j = t; j < Nn; j += 256) rowbuf[j] = 0.f;
  if (t < KCAND) evv[t] = expf(sig[row * KCAND + t]);
  __syncthreads();
  if (t == 0) {
    float ssum = 0.f;
    for (int k = 0; k < KCAND; ++k) ssum += evv[k];
    sinv = 1.0f / ssum;
  }
  __syncthreads();
  if (t < KCAND) rowbuf[cols[row * KCAND + t]] = evv[t] * sinv;
  __syncthreads();
  const float4* r4 = (const float4*)rowbuf;
  float4 v = r4[t];
#pragma unroll
  for (int b = 0; b < Bb; ++b) {
    float4* o4 = (float4*)(out + ((size_t)b * Nn + row) * Nn);
    o4[t] = v;
  }
}

// ---------------- host-side launch ----------------
extern "C" void kernel_launch(void* const* d_in, const int* in_sizes, int n_in,
                              void* d_out, int out_size, void* d_ws, size_t ws_size,
                              hipStream_t stream) {
  const float* x      = (const float*)d_in[0];
  // d_in[1] (A_original) only contributes its shape; unused for values.
  const float* srcsig = (const float*)d_in[2];
  const float* srcB   = (const float*)d_in[3];
  const float* w1     = (const float*)d_in[4];
  const float* b1     = (const float*)d_in[5];
  const float* w2     = (const float*)d_in[6];
  const float* b2     = (const float*)d_in[7];
  float* out = (float*)d_out;

  char* p = (char*)d_ws;
  auto alloc = [&](size_t bytes) -> char* {
    char* q = p;
    p += (bytes + 255) & ~(size_t)255;
    return q;
  };
  float*    s_    = (float*)alloc((size_t)Nn * Tt * 4);
  float*    sn_   = (float*)alloc((size_t)Nn * Tt * 4);
  float*    mu_   = (float*)alloc(Nn * 4);
  float*    sd_   = (float*)alloc(Nn * 4);
  float*    nrm_  = (float*)alloc(Nn * 4);
  _Float16* snh_  = (_Float16*)alloc((size_t)Nn * Tt * 2);
  _Float16* ph_   = (_Float16*)alloc((size_t)Nn * Tt * 2);
  _Float16* qh_   = (_Float16*)alloc((size_t)Nn * Tt * 2);
  float*    R_    = (float*)alloc((size_t)Nn * Nn * 4);
  _Float16* Rh_   = (_Float16*)alloc((size_t)Nn * Nn * 2);
  int*      cols_ = (int*)alloc(Ee * 4);
  float*    Va_   = (float*)alloc((size_t)Nn * NEV * 4);
  float*    Vb_   = (float*)alloc((size_t)Nn * NEV * 4);
  _Float16* Vah_  = (_Float16*)alloc((size_t)Nn * NEV * 2);
  float*    Lc_   = (float*)alloc(NEV * NEV * 4);
  float*    E8_   = (float*)alloc((size_t)Nn * NMFR * 4);
  float*    lam8_ = (float*)alloc(NMFR * 4);
  int*      idx8_ = (int*)alloc(NMFR * 4);
  int*      idxw_ = (int*)alloc(NMFR * 4);
  float*    wsc_  = (float*)alloc(NMFR * 4);
  float*    WmA_  = (float*)alloc((size_t)Nn * NMFR * 4);
  float*    WmB_  = (float*)alloc((size_t)Nn * NMFR * 4);
  _Float16* WhA_  = (_Float16*)alloc((size_t)Nn * NEV * 2);
  _Float16* WhB_  = (_Float16*)alloc((size_t)Nn * NEV * 2);
  float*    Cg_   = (float*)alloc(NMFR * NMFR * 4);
  float*    U_    = (float*)alloc((size_t)Nn * NMFR * 4);
  float*    rho_  = (float*)alloc(4 * 4);
  float*    sclr_ = (float*)alloc(Ee * 4);
  float*    scro_ = (float*)alloc(Ee * 4);
  float*    scre_ = (float*)alloc(Ee * 4);
  float*    sig_  = (float*)alloc(Ee * 4);
  float*    st_   = (float*)alloc(8 * 4);
  int*      gmax_ = (int*)alloc(2 * 4);

  k_init<<<1, 32, 0, stream>>>(gmax_);
  k_prep<<<Nn, 128, 0, stream>>>(x, s_, sn_, mu_, sd_, nrm_, snh_, ph_, qh_);
  k_relation<<<dim3(Nn / 32, Nn / 64), 256, 0, stream>>>(snh_, ph_, qh_, R_, Rh_);
  k_topk<<<Nn, 32, 0, stream>>>(R_, cols_);

  // subspace iteration for top-16 eigenpairs of R
  k_einit<<<Nn / 256, 256, 0, stream>>>(Va_, Vah_);
  for (int it = 0; it < 12; ++it) {
    k_mv<<<Nn / 16, 32, 0, stream>>>(Rh_, Vah_, Vb_);
    k_gram<<<1, 256, 0, stream>>>(Vb_, Lc_);
    k_ortho<<<Nn / 256, 256, 0, stream>>>(Vb_, Lc_, Va_, Vah_);
  }
  k_mv<<<Nn / 16, 32, 0, stream>>>(Rh_, Vah_, Vb_);   // AV for Ritz values
  k_lam<<<1, 1024, 0, stream>>>(Va_, Vb_, srcsig, rho_, lam8_, idx8_, idxw_, wsc_);
  k_post<<<Nn / 256, 256, 0, stream>>>(Va_, idx8_, idxw_, wsc_, E8_, WmA_, WhA_);

  // 80 NMF gradient steps, M*W through WMMA each step
  float*    wmc = WmA_; float*    wmn = WmB_;
  _Float16* whc = WhA_; _Float16* whn = WhB_;
  for (int it = 0; it < 80; ++it) {
    k_nmf_gram<<<1, 64, 0, stream>>>(wmc, Cg_);
    k_nmf_step<<<Nn / 16, 32, 0, stream>>>(Rh_, whc, wmc, Cg_, wmn, whn);
    float* tf = wmc; wmc = wmn; wmn = tf;
    _Float16* th = whc; whc = whn; whn = th;
  }
  // 80 is even: final W is back in WmA_ (== wmc)

  k_roleU<<<(Nn * NMFR + 255) / 256, 256, 0, stream>>>(wmc, rho_, srcB, U_);
  k_pairmax<<<(Nn * Nn) / 256, 256, 0, stream>>>(E8_, E8_, lam8_, gmax_, 0);
  k_pairmax<<<(Nn * Nn) / 256, 256, 0, stream>>>(U_, wmc, nullptr, gmax_, 1);
  k_lrscore<<<Nn, 32, 0, stream>>>(E8_, lam8_, cols_, gmax_, sclr_);
  k_rolescore<<<(Ee + 255) / 256, 256, 0, stream>>>(U_, wmc, cols_, gmax_, scro_);
  k_resscore<<<(Ee + 255) / 256, 256, 0, stream>>>(s_, sn_, mu_, sd_, nrm_, cols_,
                                                   w1, b1, w2, b2, scre_);
  k_stats<<<1, 1024, 0, stream>>>(sclr_, scro_, scre_, st_);
  k_final<<<(Ee + 255) / 256, 256, 0, stream>>>(sclr_, scro_, scre_, st_, sig_);
  k_emit<<<Nn, 256, 0, stream>>>(sig_, cols_, out);
}
